// PoissonEquation2DPINN_60919816127005
// MI455X (gfx1250) — compile-verified, matrix operand order checked by reading, not run on hardware
//
#include <hip/hip_runtime.h>
#include <hip/hip_bf16.h>
#include <math.h>

typedef __attribute__((ext_vector_type(16))) _Float16 v16h;
typedef __attribute__((ext_vector_type(8)))  _Float16 v8h;
typedef __attribute__((ext_vector_type(8)))  float    v8f;

#define PITCH 136   // halfs per LDS row: 128 + 8 pad -> 272B rows, 16B aligned, bank-spread
#define KC    4     // K chunks of 32 -> K=128
#define PI_P  32    // interior points per block (5 channels -> 160 rows -> 10 M-tiles)
#define BD_P  128   // boundary points per block (8 M-tiles)

// ---------------------------------------------------------------------------
// Z[R x 128] = X[R x 128] @ W[128 x 128] via v_wmma_f32_16x16x32_f16.
// 8 waves: wave w owns output columns [16w, 16w+16). B fragments live in VGPRs
// for the whole call; A fragments stream from LDS (2x ds_load_b128 each).
// Wfrag layout (prep kernel): [nt][kc][lane][16] halfs -> 32B contiguous/lane.
// ---------------------------------------------------------------------------
__device__ __forceinline__ void wmma_gemm128(const _Float16* __restrict__ Wfrag,
                                             const _Float16* __restrict__ sX,
                                             _Float16* __restrict__ sZ,
                                             int nMt, int wave, int lane)
{
  v16h b[KC];
#pragma unroll
  for (int kc = 0; kc < KC; ++kc)
    b[kc] = *(const v16h*)(Wfrag + ((wave * KC + kc) * 32 + lane) * 16);

  const int mcol = lane & 15;            // A row within tile / D column within tile
  const int base = (lane < 16) ? 0 : 8;  // K sub-offset for A, M sub-offset for D

  for (int mt = 0; mt < nMt; ++mt) {
    union { v16h v; v8h h[2]; } a[KC];
    const _Float16* xrow = sX + (mt * 16 + mcol) * PITCH;
#pragma unroll
    for (int kc = 0; kc < KC; ++kc) {
      a[kc].h[0] = *(const v8h*)(xrow + kc * 32 + base);        // K = 32kc+base..+7
      a[kc].h[1] = *(const v8h*)(xrow + kc * 32 + 16 + base);   // K = 32kc+16+base..+7
    }
    v8f c = {0.f, 0.f, 0.f, 0.f, 0.f, 0.f, 0.f, 0.f};
#pragma unroll
    for (int kc = 0; kc < KC; ++kc)
      c = __builtin_amdgcn_wmma_f32_16x16x32_f16(false, a[kc].v,
                                                 false, b[kc],
                                                 (short)0, c,
                                                 false, false);
    // D layout: lane -> col n, VGPR r -> row r+base
    const int n = wave * 16 + mcol;
    _Float16* zrow = sZ + (mt * 16 + base) * PITCH + n;
#pragma unroll
    for (int r = 0; r < 8; ++r)
      zrow[r * PITCH] = (_Float16)c[r];
  }
}

// ---------------------------------------------------------------------------
// Re-tile W (128x128 row-major [k][n] f32) into WMMA fragment-major f16.
// idx = ((nt*4 + kc)*32 + lane)*16 + e
// ---------------------------------------------------------------------------
__global__ void prep_weights(const float* __restrict__ W1,
                             const float* __restrict__ W2,
                             _Float16* __restrict__ wf)
{
  int idx = blockIdx.x * blockDim.x + threadIdx.x;
  if (idx >= 8 * KC * 32 * 16) return;
  int e     = idx & 15;
  int lane  = (idx >> 4) & 31;
  int kc    = (idx >> 9) & 3;
  int n     = ((idx >> 11) << 4) + (lane & 15);
  int basek = (lane < 16) ? 0 : 8;
  int k     = kc * 32 + ((e < 8) ? (basek + e) : (16 + basek + (e - 8)));
  wf[idx]         = (_Float16)W1[k * 128 + n];
  wf[16384 + idx] = (_Float16)W2[k * 128 + n];
}

__global__ void zero_out(float* __restrict__ out)
{
  if (threadIdx.x < 2) out[threadIdx.x] = 0.f;
}

// tanh coupling for the 5 forward-mode channels (v, gx, gy, hxx, hyy)
__device__ __forceinline__ void pointwise_int(const _Float16* __restrict__ sZ,
                                              _Float16* __restrict__ sX,
                                              const float* __restrict__ bias,
                                              int tid)
{
  const int P = PI_P;
  for (int idx = tid; idx < P * 128; idx += 256) {
    int p = idx >> 7, n = idx & 127;
    float zv  = (float)sZ[(0 * P + p) * PITCH + n] + bias[n];
    float zgx = (float)sZ[(1 * P + p) * PITCH + n];
    float zgy = (float)sZ[(2 * P + p) * PITCH + n];
    float zhx = (float)sZ[(3 * P + p) * PITCH + n];
    float zhy = (float)sZ[(4 * P + p) * PITCH + n];
    float av = tanhf(zv);
    float t  = 1.f - av * av;
    sX[(0 * P + p) * PITCH + n] = (_Float16)av;
    sX[(1 * P + p) * PITCH + n] = (_Float16)(t * zgx);
    sX[(2 * P + p) * PITCH + n] = (_Float16)(t * zgy);
    sX[(3 * P + p) * PITCH + n] = (_Float16)(t * (zhx - 2.f * av * zgx * zgx));
    sX[(4 * P + p) * PITCH + n] = (_Float16)(t * (zhy - 2.f * av * zgy * zgy));
  }
}

// ---------------------------------------------------------------------------
// Interior: forward-mode Laplacian, loss_f partial -> out[1]
// ---------------------------------------------------------------------------
__global__ void pinn_interior(const float* __restrict__ xy,
                              const float* __restrict__ f,
                              const float* __restrict__ W0,
                              const float* __restrict__ b0,
                              const float* __restrict__ b1,
                              const float* __restrict__ b2,
                              const float* __restrict__ W3,
                              const _Float16* __restrict__ W1f,
                              const _Float16* __restrict__ W2f,
                              float* __restrict__ out, int n_int)
{
  extern __shared__ char smem[];
  _Float16* sX   = (_Float16*)smem;                 // [160][PITCH]
  _Float16* sZ   = sX + 5 * PI_P * PITCH;           // [160][PITCH]
  float*    sDot = (float*)(sZ + 5 * PI_P * PITCH); // [64] (hx, hy dots)
  float*    sW3  = sDot + 64;                       // [128]
  float*    sSum = sW3 + 128;

  const int tid  = threadIdx.x;
  const int wave = tid >> 5, lane = tid & 31;
  const int p0   = blockIdx.x * PI_P;

  if (tid == 0) *sSum = 0.f;
  if (tid < 128) sW3[tid] = W3[tid];
  __builtin_prefetch(W1f + (wave * KC * 32) * 16, 0, 1);

  // layer 0 (K=2): analytic value/grad/Hessian seeds, done in VALU
  for (int idx = tid; idx < PI_P * 128; idx += 256) {
    int p = idx >> 7, n = idx & 127;
    int gp = p0 + p;
    float x = 0.f, y = 0.f;
    if (gp < n_int) { x = xy[2 * gp]; y = xy[2 * gp + 1]; }
    float wx = W0[n], wy = W0[128 + n];
    float z  = fmaf(x, wx, fmaf(y, wy, b0[n]));
    float av = tanhf(z);
    float t  = 1.f - av * av;
    sX[(0 * PI_P + p) * PITCH + n] = (_Float16)av;
    sX[(1 * PI_P + p) * PITCH + n] = (_Float16)(t * wx);
    sX[(2 * PI_P + p) * PITCH + n] = (_Float16)(t * wy);
    sX[(3 * PI_P + p) * PITCH + n] = (_Float16)(-2.f * av * t * wx * wx);
    sX[(4 * PI_P + p) * PITCH + n] = (_Float16)(-2.f * av * t * wy * wy);
  }
  __syncthreads();

  // layers 1 and 2: GEMM (WMMA) + tanh coupling
  wmma_gemm128(W1f, sX, sZ, 5 * PI_P / 16, wave, lane);
  __syncthreads();
  pointwise_int(sZ, sX, b1, tid);
  __syncthreads();
  wmma_gemm128(W2f, sX, sZ, 5 * PI_P / 16, wave, lane);
  __syncthreads();
  pointwise_int(sZ, sX, b2, tid);
  __syncthreads();

  // layer 3 (N=1): only hxx/hyy rows matter for the Laplacian (b3 drops out)
  if (tid < 2 * PI_P) {
    int row = 3 * PI_P + tid;  // hx rows then hy rows
    float acc = 0.f;
#pragma unroll 8
    for (int k = 0; k < 128; ++k)
      acc += (float)sX[row * PITCH + k] * sW3[k];
    sDot[tid] = acc;
  }
  __syncthreads();
  if (tid < PI_P) {
    int gp = p0 + tid;
    if (gp < n_int) {
      float lap = sDot[tid] + sDot[PI_P + tid];
      float d   = lap - f[gp];
      atomicAdd(sSum, d * d);
    }
  }
  __syncthreads();
  if (tid == 0) atomicAdd(out + 1, (0.5f / (float)n_int) * (*sSum));
}

// ---------------------------------------------------------------------------
// Boundary: plain forward, loss_bound partial -> out[0]
// ---------------------------------------------------------------------------
__global__ void pinn_boundary(const float* __restrict__ xy,
                              const float* __restrict__ g,
                              const float* __restrict__ W0,
                              const float* __restrict__ b0,
                              const float* __restrict__ b1,
                              const float* __restrict__ b2,
                              const float* __restrict__ W3,
                              const float* __restrict__ b3,
                              const _Float16* __restrict__ W1f,
                              const _Float16* __restrict__ W2f,
                              float* __restrict__ out, int n_bd)
{
  extern __shared__ char smem[];
  _Float16* sX  = (_Float16*)smem;              // [128][PITCH]
  _Float16* sZ  = sX + BD_P * PITCH;            // [128][PITCH]
  float*    sW3 = (float*)(sZ + BD_P * PITCH);  // [128]
  float*    sSum = sW3 + 128;

  const int tid  = threadIdx.x;
  const int wave = tid >> 5, lane = tid & 31;
  const int p0   = blockIdx.x * BD_P;

  if (tid == 0) *sSum = 0.f;
  if (tid < 128) sW3[tid] = W3[tid];

  for (int idx = tid; idx < BD_P * 128; idx += 256) {
    int p = idx >> 7, n = idx & 127;
    int gp = p0 + p;
    float x = 0.f, y = 0.f;
    if (gp < n_bd) { x = xy[2 * gp]; y = xy[2 * gp + 1]; }
    float z = fmaf(x, W0[n], fmaf(y, W0[128 + n], b0[n]));
    sX[p * PITCH + n] = (_Float16)tanhf(z);
  }
  __syncthreads();

  wmma_gemm128(W1f, sX, sZ, BD_P / 16, wave, lane);
  __syncthreads();
  for (int idx = tid; idx < BD_P * 128; idx += 256) {
    int p = idx >> 7, n = idx & 127;
    sX[p * PITCH + n] = (_Float16)tanhf((float)sZ[p * PITCH + n] + b1[n]);
  }
  __syncthreads();
  wmma_gemm128(W2f, sX, sZ, BD_P / 16, wave, lane);
  __syncthreads();
  for (int idx = tid; idx < BD_P * 128; idx += 256) {
    int p = idx >> 7, n = idx & 127;
    sX[p * PITCH + n] = (_Float16)tanhf((float)sZ[p * PITCH + n] + b2[n]);
  }
  __syncthreads();

  if (tid < BD_P) {
    int gp = p0 + tid;
    if (gp < n_bd) {
      float acc = b3[0];
#pragma unroll 8
      for (int k = 0; k < 128; ++k)
        acc += (float)sX[tid * PITCH + k] * sW3[k];
      float d = acc - g[gp];
      atomicAdd(sSum, d * d);
    }
  }
  __syncthreads();
  if (tid == 0) atomicAdd(out + 0, (0.5f / (float)n_bd) * (*sSum));
}

// ---------------------------------------------------------------------------
extern "C" void kernel_launch(void* const* d_in, const int* in_sizes, int n_in,
                              void* d_out, int out_size, void* d_ws, size_t ws_size,
                              hipStream_t stream)
{
  const float* xy_int = (const float*)d_in[0];
  const float* f      = (const float*)d_in[1];
  const float* xy_bd  = (const float*)d_in[2];
  const float* g      = (const float*)d_in[3];
  const float* W0     = (const float*)d_in[4];
  const float* b0     = (const float*)d_in[5];
  const float* W1     = (const float*)d_in[6];
  const float* b1     = (const float*)d_in[7];
  const float* W2     = (const float*)d_in[8];
  const float* b2     = (const float*)d_in[9];
  const float* W3     = (const float*)d_in[10];
  const float* b3     = (const float*)d_in[11];
  float* out = (float*)d_out;

  const int n_int = in_sizes[1];
  const int n_bd  = in_sizes[3];

  _Float16* wfrag = (_Float16*)d_ws;               // W1f [16384], W2f [16384]
  const _Float16* W1f = wfrag;
  const _Float16* W2f = wfrag + 16384;

  const size_t smem_i = (size_t)(5 * PI_P * PITCH) * 2 * 2 + (64 + 128 + 1) * 4;
  const size_t smem_b = (size_t)(BD_P * PITCH) * 2 * 2 + (128 + 1) * 4;

  (void)hipFuncSetAttribute((const void*)pinn_interior,
                            hipFuncAttributeMaxDynamicSharedMemorySize, (int)smem_i);
  (void)hipFuncSetAttribute((const void*)pinn_boundary,
                            hipFuncAttributeMaxDynamicSharedMemorySize, (int)smem_b);

  zero_out<<<1, 64, 0, stream>>>(out);
  prep_weights<<<64, 256, 0, stream>>>(W1, W2, wfrag);

  int gi = (n_int + PI_P - 1) / PI_P;
  pinn_interior<<<gi, 256, smem_i, stream>>>(xy_int, f, W0, b0, b1, b2, W3,
                                             W1f, W2f, out, n_int);

  int gb = (n_bd + BD_P - 1) / BD_P;
  pinn_boundary<<<gb, 256, smem_b, stream>>>(xy_bd, g, W0, b0, b1, b2, W3, b3,
                                             W1f, W2f, out, n_bd);
}
